// ParticlePropagator_60619168416034
// MI455X (gfx1250) — compile-verified, matrix operand order checked by reading, not run on hardware
//
#include <hip/hip_runtime.h>
#include <cstdint>

// ---------------------------------------------------------------------------
// ParticlePropagator for MI455X (gfx1250, wave32, WMMA bf16)
//   N=1024 nodes, IN_DIM=6, POS2=2, D=64, H=300 (padded to 320), steps=2
// ---------------------------------------------------------------------------

typedef __attribute__((ext_vector_type(16))) __bf16 v16bf;
typedef __attribute__((ext_vector_type(8)))  float  v8f;

#define NP   1024       // particles
#define HP   320        // hidden (300) padded to 320 = 10 K-chunks of 32
#define DE   64

// ---- workspace byte offsets (contiguous fragment tables first) ----
#define WS_WR    0u          // re_w1   frags: 320x64  -> 10*4  * 1024B = 40960
#define WS_W0BC  40960u      // [W0c;W0b]    : 128x320 -> 4*20  * 1024B = 81920
#define WS_W1    122880u     // ee_w1        : 320x320 -> 10*20 * 1024B = 204800
#define WS_W2    327680u     // ee_w2        : 320x64  -> 10*4  * 1024B = 40960
#define WS_U     368640u     // u : 1024x320 f32 (coords_i @ re_w0[:2] + re_b0)
#define WS_V     1679360u    // v : 1024x320 f32 (coords_j @ re_w0[2:])
#define WS_A     2990080u    // a : 1024x320 f32 (h_i @ W0a + ee_b0)
#define WS_ENCP  4300800u    // enc_p : 1024x64 f32
#define WS_H0    4562944u    // h ping
#define WS_H1    4825088u    // h pong
#define WS_SE    5087232u    // sum_eff : 1024x64 f32
// total ~5.1 MB

#define EDGE_SMEM (204800 + 8*1024 + 8*4096)   // W1 (LDS) + bounce + reduce = 245760 B

// wave-private fence: bounce tile is per-wave; same-wave LDS ops complete
// in order (DScnt, "LDS: in order"), so only compiler ordering is needed.
#define WAVE_FENCE() asm volatile("" ::: "memory")

// ---------------------------------------------------------------------------
// helpers
// ---------------------------------------------------------------------------
__device__ __forceinline__ unsigned short f2bf(float f) {
  union { __bf16 b; unsigned short u; } c;
  c.b = (__bf16)f;              // hardware f32->bf16 convert (RNE)
  return c.u;
}

union FragB { uint4 q[2]; v16bf v; __bf16 h[16]; unsigned short us[16]; };

__device__ __forceinline__ v8f zero8() {
  v8f z = {0.f,0.f,0.f,0.f,0.f,0.f,0.f,0.f};
  return z;
}

// D = A(16x32 bf16) * B(32x16 bf16) + C(16x16 f32)
__device__ __forceinline__ v8f wmma_bf16(v16bf a, v16bf b, v8f c) {
  return __builtin_amdgcn_wmma_f32_16x16x32_bf16(false, a, false, b, (short)0, c,
                                                 false, false);
}

// fragment-linear B operand: frag = 512 bf16 = 64 uint4; lane holds 16 bf16.
__device__ __forceinline__ v16bf load_frag(const uint4* base, int frag, int lane) {
  FragB f;
  f.q[0] = base[frag * 64 + lane * 2 + 0];
  f.q[1] = base[frag * 64 + lane * 2 + 1];
  return f.v;
}

// A-fragment (lane = row, 16 consecutive K elems): relu(u[k..]+v[k..]) -> bf16
__device__ __forceinline__ v16bf relu_sum_bf16(const float* pu, const float* pv) {
  const float4* a4 = (const float4*)pu;
  const float4* b4 = (const float4*)pv;
  FragB f;
#pragma unroll
  for (int q = 0; q < 4; q++) {
    float4 xa = a4[q], xb = b4[q];
    f.h[q*4+0] = (__bf16)fmaxf(xa.x + xb.x, 0.f);
    f.h[q*4+1] = (__bf16)fmaxf(xa.y + xb.y, 0.f);
    f.h[q*4+2] = (__bf16)fmaxf(xa.z + xb.z, 0.f);
    f.h[q*4+3] = (__bf16)fmaxf(xa.w + xb.w, 0.f);
  }
  return f.v;
}

// A-fragment from 16 consecutive f32 (no activation)
__device__ __forceinline__ v16bf row_bf16(const float* p) {
  const float4* a4 = (const float4*)p;
  FragB f;
#pragma unroll
  for (int q = 0; q < 4; q++) {
    float4 xa = a4[q];
    f.h[q*4+0] = (__bf16)xa.x;
    f.h[q*4+1] = (__bf16)xa.y;
    f.h[q*4+2] = (__bf16)xa.z;
    f.h[q*4+3] = (__bf16)xa.w;
  }
  return f.v;
}

// ---------------------------------------------------------------------------
// prep: convert weights to zero-padded fragment-linear bf16 tables
//   B-frag convention: lane l -> n = nc*16 + (l&15); elem e -> k = kc*32 + (l>>4)*16 + e
// ---------------------------------------------------------------------------
__global__ void prep_weights_kernel(const float* __restrict__ re_w1,
                                    const float* __restrict__ ee_w0,
                                    const float* __restrict__ ee_w1,
                                    const float* __restrict__ ee_w2,
                                    unsigned short* __restrict__ dst) {
  const int idx = blockIdx.x * 256 + threadIdx.x;
  if (idx >= 184320) return;
  const float* W; int rel, Kt, Nt, NC, pitch; bool stack = false;
  if (idx < 20480)       { rel = idx;          W = re_w1; Kt = 300; Nt = 64;  NC = 4;  pitch = 64;  }
  else if (idx < 61440)  { rel = idx - 20480;  W = ee_w0; Kt = 128; Nt = 300; NC = 20; pitch = 300; stack = true; }
  else if (idx < 163840) { rel = idx - 61440;  W = ee_w1; Kt = 300; Nt = 300; NC = 20; pitch = 300; }
  else                   { rel = idx - 163840; W = ee_w2; Kt = 300; Nt = 64;  NC = 4;  pitch = 64;  }
  const int frag = rel >> 9, within = rel & 511, l = within >> 4, e = within & 15;
  const int kc = frag / NC, nc = frag % NC;
  const int k = kc * 32 + ((l >> 4) << 4) + e;
  const int n = nc * 16 + (l & 15);
  float val = 0.f;
  if (k < Kt && n < Nt) {
    // stacked [W0c; W0b]: rows 0..63 <- ee_w0 rows 128..191 (enc_r), 64..127 <- rows 64..127 (h_j)
    const int sk = stack ? ((k < 64) ? (k + 128) : k) : k;
    val = W[sk * pitch + n];
  }
  dst[idx] = f2bf(val);
}

// ---------------------------------------------------------------------------
// node encoders: enc_p = mlp2(x), u/v rank-2 tables for enc_r layer0, h0 = 0
// ---------------------------------------------------------------------------
__global__ void encode_nodes_kernel(const float* __restrict__ x,
                                    const float* __restrict__ pe_w0, const float* __restrict__ pe_b0,
                                    const float* __restrict__ pe_w1, const float* __restrict__ pe_b1,
                                    const float* __restrict__ re_w0, const float* __restrict__ re_b0,
                                    float* __restrict__ encp, float* __restrict__ u,
                                    float* __restrict__ v, float* __restrict__ h0) {
  __shared__ float hid[304];
  __shared__ float xr[6];
  const int n = blockIdx.x, tid = threadIdx.x;
  if (tid < 6) xr[tid] = x[n * 6 + tid];
  __syncthreads();
  for (int k = tid; k < 300; k += 256) {
    float s = pe_b0[k];
#pragma unroll
    for (int d = 0; d < 6; d++) s += xr[d] * pe_w0[d * 300 + k];
    hid[k] = fmaxf(s, 0.f);
  }
  for (int k = tid; k < HP; k += 256) {
    float uu = 0.f, vv = 0.f;
    if (k < 300) {
      uu = xr[0] * re_w0[k]       + xr[1] * re_w0[300 + k] + re_b0[k];
      vv = xr[0] * re_w0[600 + k] + xr[1] * re_w0[900 + k];
    }
    u[n * HP + k] = uu;
    v[n * HP + k] = vv;
  }
  __syncthreads();
  for (int o = tid; o < DE; o += 256) {
    float s = pe_b1[o];
    for (int k = 0; k < 300; k++) s += hid[k] * pe_w1[k * DE + o];
    encp[n * DE + o] = fmaxf(s, 0.f);
    h0[n * DE + o] = 0.f;
  }
}

// per-step: a[i] = h[i] @ W0a + ee_b0 ; zero sum_eff
__global__ void node_pre_kernel(const float* __restrict__ h,
                                const float* __restrict__ ee_w0,
                                const float* __restrict__ ee_b0,
                                float* __restrict__ a, float* __restrict__ se) {
  __shared__ float hr[64];
  const int n = blockIdx.x, tid = threadIdx.x;
  if (tid < 64) { hr[tid] = h[n * DE + tid]; se[n * DE + tid] = 0.f; }
  __syncthreads();
  for (int k = tid; k < HP; k += 256) {
    float s = 0.f;
    if (k < 300) {
      s = ee_b0[k];
      for (int d = 0; d < 64; d++) s += hr[d] * ee_w0[d * 300 + k];
    }
    a[n * HP + k] = s;
  }
}

// ---------------------------------------------------------------------------
// THE edge kernel: recompute enc_r, run the 3-layer edge MLP with bf16 WMMA,
// reduce over i, atomically accumulate sum_eff[j].
//   grid (16, 64): blockIdx.x -> 64 i's (8 waves x 8 iters), blockIdx.y -> 16 j's
// ---------------------------------------------------------------------------
__global__ __launch_bounds__(256) void edge_kernel(
    const float* __restrict__ ubuf, const float* __restrict__ vbuf,
    const float* __restrict__ abuf, const float* __restrict__ hbuf,
    const uint4* __restrict__ wrF, const uint4* __restrict__ w0bcF,
    const uint4* __restrict__ w1F, const uint4* __restrict__ w2F,
    const float* __restrict__ re_b1, const float* __restrict__ ee_b1,
    const float* __restrict__ ee_b2, float* __restrict__ sum_eff) {
  extern __shared__ char smem[];
  uint4* w1s = (uint4*)smem;                              // 204800 B = 12800 uint4
  const int tid = threadIdx.x, wv = tid >> 5, lane = tid & 31;
  const int hf = lane >> 4, lm = lane & 15;
  const int j0 = blockIdx.y * 16, jrow = j0 + lm;

  // stage W1 fragment table into LDS (L2 resident -> one-time 200KB fill)
  for (int idx = tid; idx < 12800; idx += 256) w1s[idx] = w1F[idx];
  __syncthreads();

  unsigned short* bS = (unsigned short*)(smem + 204800 + wv * 1024);   // 16x32 bf16 bounce
  const uint4*    b4 = (const uint4*)(smem + 204800 + wv * 1024);
  float*         red = (float*)(smem + 204800 + 8192);                 // 8 waves x 16x64 f32

  // h_j as A-fragments (K = 0..63), constant over the i-loop
  v16bf hjA[2];
#pragma unroll
  for (int c = 0; c < 2; c++) hjA[c] = row_bf16(hbuf + jrow * DE + c * 32 + hf * 16);

  v8f sumacc[4];
#pragma unroll
  for (int t = 0; t < 4; t++) sumacc[t] = zero8();

  for (int it = 0; it < 8; ++it) {
    const int i = blockIdx.x * 64 + wv * 8 + it;

    // ---- t0 = relu(u[i] + v[j])  (16 x 320 A-fragments) ----
    v16bf zA[10];
#pragma unroll
    for (int kc = 0; kc < 10; kc++) {
      const int K0 = kc * 32 + hf * 16;
      zA[kc] = relu_sum_bf16(ubuf + i * HP + K0, vbuf + jrow * HP + K0);
    }

    // ---- enc_r = relu(t0 @ re_w1 + re_b1)  (16 x 64) ----
    v16bf encA[2];
#pragma unroll
    for (int p = 0; p < 2; p++) {
#pragma unroll
      for (int sub = 0; sub < 2; sub++) {
        const int nc = p * 2 + sub;
        // two independent accumulator chains: consecutive WMMAs are
        // independent -> no WMMA->WMMA hazard NOPs (ISA 7.12.1)
        v8f acc0 = zero8(), acc1 = zero8();
        v16bf bcur = load_frag(wrF, 0 * 4 + nc, lane);     // double-buffered B
#pragma unroll
        for (int kc = 0; kc < 10; kc++) {
          const int kn = (kc + 1 < 10) ? kc + 1 : kc;
          v16bf bnext = load_frag(wrF, kn * 4 + nc, lane);
          if (kc & 1) acc1 = wmma_bf16(zA[kc], bcur, acc1);
          else        acc0 = wmma_bf16(zA[kc], bcur, acc0);
          bcur = bnext;
        }
        const float bb = re_b1[nc * 16 + lm];
#pragma unroll
        for (int r = 0; r < 8; r++)
          bS[(r + hf * 8) * 32 + sub * 16 + lm] =
              f2bf(fmaxf(acc0[r] + acc1[r] + bb, 0.f));
      }
      WAVE_FENCE();   // per-wave bounce: same-wave LDS ops are in order
      { FragB f; f.q[0] = b4[lm * 4 + hf * 2]; f.q[1] = b4[lm * 4 + hf * 2 + 1]; encA[p] = f.v; }
      WAVE_FENCE();
    }

    // ---- z1 = relu([enc_r | h_j] @ [W0c;W0b] + a[i])  (16 x 320) ----
    for (int p = 0; p < 10; p++) {
#pragma unroll
      for (int sub = 0; sub < 2; sub++) {
        const int nc = p * 2 + sub;
        // issue all 4 B-frag loads up front, then two independent WMMA chains
        v16bf bf0 = load_frag(w0bcF, 0 * 20 + nc, lane);
        v16bf bf1 = load_frag(w0bcF, 1 * 20 + nc, lane);
        v16bf bf2 = load_frag(w0bcF, 2 * 20 + nc, lane);
        v16bf bf3 = load_frag(w0bcF, 3 * 20 + nc, lane);
        v8f acc0 = zero8(), acc1 = zero8();
        acc0 = wmma_bf16(encA[0], bf0, acc0);
        acc1 = wmma_bf16(encA[1], bf1, acc1);
        acc0 = wmma_bf16(hjA[0],  bf2, acc0);
        acc1 = wmma_bf16(hjA[1],  bf3, acc1);
        const float av = abuf[i * HP + nc * 16 + lm];   // h_i@W0a + ee_b0 (0 in pad)
#pragma unroll
        for (int r = 0; r < 8; r++)
          bS[(r + hf * 8) * 32 + sub * 16 + lm] =
              f2bf(fmaxf(acc0[r] + acc1[r] + av, 0.f));
      }
      WAVE_FENCE();
      { FragB f; f.q[0] = b4[lm * 4 + hf * 2]; f.q[1] = b4[lm * 4 + hf * 2 + 1]; zA[p] = f.v; }
      WAVE_FENCE();
    }

    // ---- z2 = relu(z1 @ W1 + b1); z3 += z2 @ W2 ----
    v8f z3acc[4];
#pragma unroll
    for (int t = 0; t < 4; t++) z3acc[t] = zero8();
    for (int p = 0; p < 10; p++) {
#pragma unroll
      for (int sub = 0; sub < 2; sub++) {
        const int nc = p * 2 + sub;
        v8f acc0 = zero8(), acc1 = zero8();
        v16bf bcur = load_frag(w1s, 0 * 20 + nc, lane);    // double-buffered B (LDS)
#pragma unroll
        for (int kc = 0; kc < 10; kc++) {
          const int kn = (kc + 1 < 10) ? kc + 1 : kc;
          v16bf bnext = load_frag(w1s, kn * 20 + nc, lane);
          if (kc & 1) acc1 = wmma_bf16(zA[kc], bcur, acc1);
          else        acc0 = wmma_bf16(zA[kc], bcur, acc0);
          bcur = bnext;
        }
        const int n = nc * 16 + lm;
        const float bb = (n < 300) ? ee_b1[n] : 0.f;
#pragma unroll
        for (int r = 0; r < 8; r++)
          bS[(r + hf * 8) * 32 + sub * 16 + lm] =
              f2bf(fmaxf(acc0[r] + acc1[r] + bb, 0.f));
      }
      WAVE_FENCE();
      v16bf z2A;
      { FragB f; f.q[0] = b4[lm * 4 + hf * 2]; f.q[1] = b4[lm * 4 + hf * 2 + 1]; z2A = f.v; }
      WAVE_FENCE();
      v16bf wb0 = load_frag(w2F, p * 4 + 0, lane);
      v16bf wb1 = load_frag(w2F, p * 4 + 1, lane);
      v16bf wb2 = load_frag(w2F, p * 4 + 2, lane);
      v16bf wb3 = load_frag(w2F, p * 4 + 3, lane);
      // 4 independent accumulators -> no hazard NOPs
      z3acc[0] = wmma_bf16(z2A, wb0, z3acc[0]);
      z3acc[1] = wmma_bf16(z2A, wb1, z3acc[1]);
      z3acc[2] = wmma_bf16(z2A, wb2, z3acc[2]);
      z3acc[3] = wmma_bf16(z2A, wb3, z3acc[3]);
    }

    // ---- eff = relu(z3 + b2); accumulate over i within this wave ----
#pragma unroll
    for (int nc2 = 0; nc2 < 4; nc2++) {
      const float bb = ee_b2[nc2 * 16 + lm];
#pragma unroll
      for (int r = 0; r < 8; r++)
        sumacc[nc2][r] += fmaxf(z3acc[nc2][r] + bb, 0.f);
    }
  }

  // ---- cross-wave reduction in LDS, then one atomic per (j, d) ----
#pragma unroll
  for (int nc2 = 0; nc2 < 4; nc2++)
#pragma unroll
    for (int r = 0; r < 8; r++)
      red[wv * 1024 + (r + hf * 8) * 64 + nc2 * 16 + lm] = sumacc[nc2][r];
  __syncthreads();
  for (int idx = tid; idx < 1024; idx += 256) {
    float s = 0.f;
#pragma unroll
    for (int w = 0; w < 8; w++) s += red[w * 1024 + idx];
    atomicAdd(&sum_eff[j0 * DE + idx], s);   // idx = m*64 + d
  }
}

// ---------------------------------------------------------------------------
// node aggregation MLP: h_out = mlp3([enc_p, sum_eff, h_in], ae_*)
// ---------------------------------------------------------------------------
__global__ void node_update_kernel(const float* __restrict__ encp,
                                   const float* __restrict__ se,
                                   const float* __restrict__ hin,
                                   const float* __restrict__ w0, const float* __restrict__ b0,
                                   const float* __restrict__ w1, const float* __restrict__ b1,
                                   const float* __restrict__ w2, const float* __restrict__ b2,
                                   float* __restrict__ hout) {
  __shared__ float inb[192];
  __shared__ float t1[300];
  __shared__ float t2[300];
  const int n = blockIdx.x, tid = threadIdx.x;
  if (tid < 64) {
    inb[tid]       = encp[n * DE + tid];
    inb[64 + tid]  = se[n * DE + tid];
    inb[128 + tid] = hin[n * DE + tid];
  }
  __syncthreads();
  for (int k = tid; k < 300; k += 256) {
    float s = b0[k];
    for (int d = 0; d < 192; d++) s += inb[d] * w0[d * 300 + k];
    t1[k] = fmaxf(s, 0.f);
  }
  __syncthreads();
  for (int k = tid; k < 300; k += 256) {
    float s = b1[k];
    for (int d = 0; d < 300; d++) s += t1[d] * w1[d * 300 + k];
    t2[k] = fmaxf(s, 0.f);
  }
  __syncthreads();
  for (int o = tid; o < DE; o += 256) {
    float s = b2[o];
    for (int d = 0; d < 300; d++) s += t2[d] * w2[d * DE + o];
    hout[n * DE + o] = fmaxf(s, 0.f);
  }
}

// decoder: predict = celu([enc_p, h] @ pd_w0 + pd_b0) @ pd_w1 + pd_b1 + x
__global__ void decode_kernel(const float* __restrict__ encp,
                              const float* __restrict__ h,
                              const float* __restrict__ x,
                              const float* __restrict__ w0, const float* __restrict__ b0,
                              const float* __restrict__ w1, const float* __restrict__ b1,
                              float* __restrict__ out) {
  __shared__ float inb[128];
  __shared__ float t1[300];
  const int n = blockIdx.x, tid = threadIdx.x;
  if (tid < 64) {
    inb[tid]      = encp[n * DE + tid];
    inb[64 + tid] = h[n * DE + tid];
  }
  __syncthreads();
  for (int k = tid; k < 300; k += 256) {
    float s = b0[k];
    for (int d = 0; d < 128; d++) s += inb[d] * w0[d * 300 + k];
    t1[k] = (s > 0.f) ? s : (expf(s) - 1.f);   // celu, alpha=1
  }
  __syncthreads();
  for (int o = tid; o < 6; o += 256) {
    float s = b1[o];
    for (int d = 0; d < 300; d++) s += t1[d] * w1[d * 6 + o];
    out[n * 6 + o] = s + x[n * 6 + o];
  }
}

// ---------------------------------------------------------------------------
extern "C" void kernel_launch(void* const* d_in, const int* in_sizes, int n_in,
                              void* d_out, int out_size, void* d_ws, size_t ws_size,
                              hipStream_t stream) {
  (void)in_sizes; (void)n_in; (void)out_size; (void)ws_size;
  const float* x     = (const float*)d_in[0];
  const float* pe_w0 = (const float*)d_in[1];
  const float* pe_b0 = (const float*)d_in[2];
  const float* pe_w1 = (const float*)d_in[3];
  const float* pe_b1 = (const float*)d_in[4];
  const float* re_w0 = (const float*)d_in[5];
  const float* re_b0 = (const float*)d_in[6];
  const float* re_w1 = (const float*)d_in[7];
  const float* re_b1 = (const float*)d_in[8];
  const float* ee_w0 = (const float*)d_in[9];
  const float* ee_b0 = (const float*)d_in[10];
  const float* ee_w1 = (const float*)d_in[11];
  const float* ee_b1 = (const float*)d_in[12];
  const float* ee_w2 = (const float*)d_in[13];
  const float* ee_b2 = (const float*)d_in[14];
  const float* ae_w0 = (const float*)d_in[15];
  const float* ae_b0 = (const float*)d_in[16];
  const float* ae_w1 = (const float*)d_in[17];
  const float* ae_b1 = (const float*)d_in[18];
  const float* ae_w2 = (const float*)d_in[19];
  const float* ae_b2 = (const float*)d_in[20];
  const float* pd_w0 = (const float*)d_in[21];
  const float* pd_b0 = (const float*)d_in[22];
  const float* pd_w1 = (const float*)d_in[23];
  const float* pd_b1 = (const float*)d_in[24];
  // d_in[25] = steps; fixed to 2 by setup_inputs (device scalar unreadable during capture)

  char* ws = (char*)d_ws;
  const uint4* wrF   = (const uint4*)(ws + WS_WR);
  const uint4* w0bcF = (const uint4*)(ws + WS_W0BC);
  const uint4* w1F   = (const uint4*)(ws + WS_W1);
  const uint4* w2F   = (const uint4*)(ws + WS_W2);
  float* ubuf = (float*)(ws + WS_U);
  float* vbuf = (float*)(ws + WS_V);
  float* abuf = (float*)(ws + WS_A);
  float* encp = (float*)(ws + WS_ENCP);
  float* h0   = (float*)(ws + WS_H0);
  float* h1   = (float*)(ws + WS_H1);
  float* se   = (float*)(ws + WS_SE);

  hipFuncSetAttribute((const void*)edge_kernel,
                      hipFuncAttributeMaxDynamicSharedMemorySize, EDGE_SMEM);

  prep_weights_kernel<<<720, 256, 0, stream>>>(re_w1, ee_w0, ee_w1, ee_w2,
                                               (unsigned short*)ws);
  encode_nodes_kernel<<<NP, 256, 0, stream>>>(x, pe_w0, pe_b0, pe_w1, pe_b1,
                                              re_w0, re_b0, encp, ubuf, vbuf, h0);

  for (int s = 0; s < 2; s++) {   // steps == 2
    const float* hi = (s == 0) ? h0 : h1;
    float*       ho = (s == 0) ? h1 : h0;
    node_pre_kernel<<<NP, 256, 0, stream>>>(hi, ee_w0, ee_b0, abuf, se);
    edge_kernel<<<dim3(16, 64), 256, EDGE_SMEM, stream>>>(
        ubuf, vbuf, abuf, hi, wrF, w0bcF, w1F, w2F, re_b1, ee_b1, ee_b2, se);
    node_update_kernel<<<NP, 256, 0, stream>>>(encp, se, hi, ae_w0, ae_b0,
                                               ae_w1, ae_b1, ae_w2, ae_b2, ho);
  }

  decode_kernel<<<NP, 256, 0, stream>>>(encp, h0, x, pd_w0, pd_b0, pd_w1, pd_b1,
                                        (float*)d_out);
}
